// SaProtLayer_31782757990834
// MI455X (gfx1250) — matrix operand order repaired and checked by reading, not compiled, not run on hardware
//
#include <hip/hip_runtime.h>

// ---------------- problem constants ----------------
#define DMODEL 1280
#define NHEAD  20
#define HDIM   64
#define FFDIM  5120
#define SEQ    1024
#define NTOK   4096   // 4 * 1024

// ---------------- vector types ----------------
typedef __attribute__((ext_vector_type(16))) __bf16       bf16x16;
typedef __attribute__((ext_vector_type(8)))  __bf16       bf16x8;
typedef __attribute__((ext_vector_type(8)))  float        f32x8;
typedef __attribute__((ext_vector_type(4)))  unsigned int u32x4;
typedef int v4i __attribute__((vector_size(16)));  // matches builtin's V4i param

// ---------------- CDNA5 async global->LDS path (guarded) ----------------
#if defined(__has_builtin)
#if __has_builtin(__builtin_amdgcn_global_load_async_to_lds_b128) && \
    __has_builtin(__builtin_amdgcn_s_wait_asynccnt)
#define USE_ASYNC_LDS 1
#endif
#endif

// copy 16 bytes global -> LDS; async (ASYNCcnt) when available.
__device__ __forceinline__ void ld16_to_lds(const void* g, void* l) {
#if defined(USE_ASYNC_LDS)
  __builtin_amdgcn_global_load_async_to_lds_b128(
      (__attribute__((address_space(1))) v4i*)(void*)g,
      (__attribute__((address_space(3))) v4i*)l, 0, 0);
#else
  *(u32x4*)l = *(const u32x4*)g;
#endif
}

__device__ __forceinline__ void async_wait0() {
#if defined(USE_ASYNC_LDS)
  __builtin_amdgcn_s_wait_asynccnt(0);
#endif
}

__device__ __forceinline__ f32x8 zero8() {
  f32x8 z = {0.f,0.f,0.f,0.f,0.f,0.f,0.f,0.f};
  return z;
}

// D = A(16x32 bf16) * B(32x16 bf16) + C(16x16 f32)
__device__ __forceinline__ f32x8 wmma_bf16(bf16x16 a, bf16x16 b, f32x8 c) {
  return __builtin_amdgcn_wmma_f32_16x16x32_bf16(false, a, false, b,
                                                 (short)0, c, false, false);
}

// A-matrix 16x32 bf16 layout (ISA 7.12.2): lane m (and m+16) holds row m;
// K chunks are two contiguous 8-element runs at kh*8 and kh*8+16 (kh = lane>>4).
__device__ __forceinline__ bf16x16 pack_a(bf16x8 lo, bf16x8 hi) {
  union { bf16x16 v; bf16x8 h[2]; } u;
  u.h[0] = lo; u.h[1] = hi;
  return u.v;
}

// ---------------- fp32 -> bf16 convert ----------------
__global__ __launch_bounds__(256) void cvt_bf16_kernel(const float* __restrict__ in,
                                                       __bf16* __restrict__ out, int n) {
  int i = blockIdx.x * 256 + threadIdx.x;
  if (i < n) out[i] = (__bf16)in[i];
}

// ---------------- layernorm (1280 cols) -> bf16 ----------------
__global__ __launch_bounds__(256) void ln_bf16_kernel(const float* __restrict__ x,
                                                      const float* __restrict__ w,
                                                      const float* __restrict__ bias,
                                                      __bf16* __restrict__ out) {
  const int row = blockIdx.x;
  const float* xr = x + (size_t)row * DMODEL;
  const int t = threadIdx.x, lane = t & 31, wv = t >> 5;
  float v[5];
  float s = 0.f;
#pragma unroll
  for (int i = 0; i < 5; ++i) { v[i] = xr[t + i * 256]; s += v[i]; }
  __shared__ float red[8];
#pragma unroll
  for (int off = 16; off; off >>= 1) s += __shfl_xor(s, off, 32);
  if (lane == 0) red[wv] = s;
  __syncthreads();
  float tot = 0.f;
#pragma unroll
  for (int i = 0; i < 8; ++i) tot += red[i];
  const float mean = tot * (1.f / DMODEL);
  float s2 = 0.f;
#pragma unroll
  for (int i = 0; i < 5; ++i) { float d = v[i] - mean; s2 += d * d; }
#pragma unroll
  for (int off = 16; off; off >>= 1) s2 += __shfl_xor(s2, off, 32);
  __syncthreads();
  if (lane == 0) red[wv] = s2;
  __syncthreads();
  float tot2 = 0.f;
#pragma unroll
  for (int i = 0; i < 8; ++i) tot2 += red[i];
  const float rstd = rsqrtf(tot2 * (1.f / DMODEL) + 1e-5f);
#pragma unroll
  for (int i = 0; i < 5; ++i) {
    const int c = t + i * 256;
    out[(size_t)row * DMODEL + c] = (__bf16)((v[i] - mean) * rstd * w[c] + bias[c]);
  }
}

// ---------------- bf16 WMMA GEMM:  C[M,N] = A[M,K] @ W[N,K]^T + bias ----------------
// 128x128 tile / block, 256 threads = 8 waves, each wave -> 32(M) x 64(N).
template<int OUT_BF16, int GELU_, int QKV, int RES>
__global__ __launch_bounds__(256) void gemm_kernel(const __bf16* __restrict__ A,
                                                   const __bf16* __restrict__ W,
                                                   const float* __restrict__ bias,
                                                   const float* __restrict__ residual,
                                                   void* __restrict__ outp,
                                                   int N, int K) {
  __shared__ __attribute__((aligned(32))) __bf16 lA[2][128 * 32];
  __shared__ __attribute__((aligned(32))) __bf16 lB[2][128 * 32];
  const int t = threadIdx.x;
  const int lane = t & 31, wv = t >> 5;
  const int wm = (wv & 3) * 32;   // wave M offset inside tile
  const int wn = (wv >> 2) * 64;  // wave N offset inside tile
  const int n0 = blockIdx.x * 128;
  const int m0 = blockIdx.y * 128;

  // stage a 128x32 bf16 panel of A and of W into LDS (async when available)
  auto stage = [&](int buf, int k0) {
#pragma unroll
    for (int i = 0; i < 2; ++i) {
      const int c = t + i * 256;           // 512 x 16B chunks
      const int r = c >> 2, off = (c & 3) * 8;
      ld16_to_lds(&A[(size_t)(m0 + r) * K + k0 + off], &lA[buf][r * 32 + off]);
      ld16_to_lds(&W[(size_t)(n0 + r) * K + k0 + off], &lB[buf][r * 32 + off]);
    }
  };

  f32x8 acc[2][4];
#pragma unroll
  for (int mt = 0; mt < 2; ++mt)
#pragma unroll
    for (int nt = 0; nt < 4; ++nt) acc[mt][nt] = zero8();

  stage(0, 0);
  async_wait0();
  __syncthreads();
  const int nk = K >> 5;
  const int rowA = lane & 15, kh = lane >> 4;
  for (int kk = 0; kk < nk; ++kk) {
    const int buf = kk & 1;
    if (kk + 1 < nk) stage(buf ^ 1, (kk + 1) << 5);
    if (kk + 2 < nk) {  // pull panels 2 steps ahead into GL2 (global_prefetch_b8)
      __builtin_prefetch(&A[(size_t)(m0 + (t >> 1)) * K + ((kk + 2) << 5)], 0, 1);
      __builtin_prefetch(&W[(size_t)(n0 + (t >> 1)) * K + ((kk + 2) << 5)], 0, 1);
    }
    bf16x16 af[2];
#pragma unroll
    for (int mt = 0; mt < 2; ++mt) {
      const __bf16* base = &lA[buf][(wm + mt * 16 + rowA) * 32];
      af[mt] = pack_a(*(const bf16x8*)&base[kh * 8], *(const bf16x8*)&base[kh * 8 + 16]);
    }
    bf16x16 bfr[4];
#pragma unroll
    for (int nt = 0; nt < 4; ++nt)   // B col = row of W: K runs at kh*8 and kh*8+16
      bfr[nt] = pack_a(*(const bf16x8*)&lB[buf][(wn + nt * 16 + rowA) * 32 + kh * 8], *(const bf16x8*)&lB[buf][(wn + nt * 16 + rowA) * 32 + kh * 8 + 16]);
#pragma unroll
    for (int mt = 0; mt < 2; ++mt)
#pragma unroll
      for (int nt = 0; nt < 4; ++nt)
        acc[mt][nt] = wmma_bf16(af[mt], bfr[nt], acc[mt][nt]);
    async_wait0();
    __syncthreads();
  }

  // C/D layout: VGPR r, lane l -> row = r + 8*(l>>4), col = l&15
  const int lrow = (lane >> 4) * 8, lcol = lane & 15;
#pragma unroll
  for (int mt = 0; mt < 2; ++mt)
#pragma unroll
    for (int nt = 0; nt < 4; ++nt) {
      const int col = n0 + wn + nt * 16 + lcol;
      const float bv = bias[col];
#pragma unroll
      for (int r = 0; r < 8; ++r) {
        const int row = m0 + wm + mt * 16 + lrow + r;
        float vv = acc[mt][nt][r] + bv;
        if (GELU_) vv = 0.5f * vv * (1.f + erff(vv * 0.70710678118f));
        if (RES) vv += residual[(size_t)row * N + col];
        if (QKV) {  // scatter into [b, H, n, 64] head-major bf16
          const int bb = row >> 10, nn = row & 1023;
          const int hh = col >> 6, dd = col & 63;
          ((__bf16*)outp)[((((size_t)bb * NHEAD + hh) << 10) + nn) * HDIM + dd] = (__bf16)vv;
        } else if (OUT_BF16) {
          ((__bf16*)outp)[(size_t)row * N + col] = (__bf16)vv;
        } else {
          ((float*)outp)[(size_t)row * N + col] = vv;
        }
      }
    }
}

// ---------------- flash attention, head dim 64, bf16 WMMA ----------------
// grid = (b*H, n/64); 128 threads = 4 waves; each wave owns 16 query rows.
__global__ __launch_bounds__(128) void attn_kernel(const __bf16* __restrict__ q,
                                                   const __bf16* __restrict__ k,
                                                   const __bf16* __restrict__ v,
                                                   const unsigned char* __restrict__ mask,
                                                   __bf16* __restrict__ out) {
  __shared__ __attribute__((aligned(32))) __bf16 lK[32 * 64];    // 32 keys x 64 dims
  __shared__ __attribute__((aligned(32))) __bf16 lVt[64 * 32];   // transposed: dims x keys
  __shared__ __attribute__((aligned(32))) __bf16 lP[4][16 * 32]; // per-wave P staging
  const int bh = blockIdx.x;
  const int b = bh / NHEAD, h = bh % NHEAD;
  const int q0 = blockIdx.y * 64;
  const int t = threadIdx.x, lane = t & 31, wv = t >> 5;
  const int lcol = lane & 15, kh = lane >> 4, lrow = kh * 8;

  const size_t headoff = (size_t)bh * SEQ * HDIM;
  const __bf16* qr = q + headoff + (size_t)(q0 + wv * 16 + lcol) * HDIM;
  bf16x16 aq[2];
#pragma unroll
  for (int c = 0; c < 2; ++c)
    aq[c] = pack_a(*(const bf16x8*)&qr[c * 32 + kh * 8],
                   *(const bf16x8*)&qr[c * 32 + kh * 8 + 16]);

  f32x8 oacc[4];
#pragma unroll
  for (int nt = 0; nt < 4; ++nt) oacc[nt] = zero8();
  float m_i[8], l_i[8];
#pragma unroll
  for (int r = 0; r < 8; ++r) { m_i[r] = -3.0e38f; l_i[r] = 0.f; }
  const float scale = 0.125f;  // 64^-0.5
  const unsigned char* mrow = mask + (size_t)b * SEQ;

  for (int kb = 0; kb < SEQ; kb += 32) {
    __syncthreads();  // previous iteration done reading lK/lVt
    {
      const __bf16* kg = k + headoff + (size_t)kb * HDIM;
      const __bf16* vg = v + headoff + (size_t)kb * HDIM;
#pragma unroll
      for (int i = 0; i < 2; ++i) {
        const int c = t + i * 128;         // 256 x 16B chunks
        const int r = c >> 3, off = (c & 7) * 8;
        ld16_to_lds(&kg[r * 64 + off], &lK[r * 64 + off]);  // K: async direct to LDS
        bf16x8 tv = *(const bf16x8*)&vg[r * 64 + off];       // V: manual transpose
#pragma unroll
        for (int j = 0; j < 8; ++j) lVt[(off + j) * 32 + r] = tv[j];
      }
    }
    async_wait0();
    __syncthreads();

    // S = Q @ K^T : two 16x16 tiles covering 32 keys
    float sreg[2][8];
#pragma unroll
    for (int j = 0; j < 2; ++j) {
      f32x8 s = zero8();
#pragma unroll
      for (int c = 0; c < 2; ++c) {
        bf16x16 bk = pack_a(*(const bf16x8*)&lK[(j * 16 + lcol) * 64 + c * 32 + kh * 8], *(const bf16x8*)&lK[(j * 16 + lcol) * 64 + c * 32 + kh * 8 + 16]);
        s = wmma_bf16(aq[c], bk, s);
      }
      const bool ok = mrow[kb + j * 16 + lcol] != 0;
#pragma unroll
      for (int r = 0; r < 8; ++r) sreg[j][r] = ok ? s[r] * scale : -3.0e38f;
    }

    // online softmax (rows live across the 16 lanes of each wave-half)
    float alpha[8];
#pragma unroll
    for (int r = 0; r < 8; ++r) {
      float mx = fmaxf(sreg[0][r], sreg[1][r]);
      mx = fmaxf(mx, __shfl_xor(mx, 1, 32));
      mx = fmaxf(mx, __shfl_xor(mx, 2, 32));
      mx = fmaxf(mx, __shfl_xor(mx, 4, 32));
      mx = fmaxf(mx, __shfl_xor(mx, 8, 32));
      const float mnew = fmaxf(m_i[r], mx);
      const float a = __expf(m_i[r] - mnew);
      const float p0 = __expf(sreg[0][r] - mnew);
      const float p1 = __expf(sreg[1][r] - mnew);
      sreg[0][r] = p0; sreg[1][r] = p1;
      float rs = p0 + p1;
      rs += __shfl_xor(rs, 1, 32);
      rs += __shfl_xor(rs, 2, 32);
      rs += __shfl_xor(rs, 4, 32);
      rs += __shfl_xor(rs, 8, 32);
      l_i[r] = l_i[r] * a + rs;
      m_i[r] = mnew;
      alpha[r] = a;
    }

    // reshape P (C-layout) -> A-layout through wave-private LDS
#pragma unroll
    for (int j = 0; j < 2; ++j)
#pragma unroll
      for (int r = 0; r < 8; ++r)
        lP[wv][(lrow + r) * 32 + j * 16 + lcol] = (__bf16)sreg[j][r];
    asm volatile("s_wait_dscnt 0x0" ::: "memory");  // wave's LDS stores visible
    const __bf16* pb = &lP[wv][lcol * 32];
    bf16x16 ap = pack_a(*(const bf16x8*)&pb[kh * 8], *(const bf16x8*)&pb[kh * 8 + 16]);

    // O = alpha*O + P @ V
#pragma unroll
    for (int nt = 0; nt < 4; ++nt) {
#pragma unroll
      for (int r = 0; r < 8; ++r) oacc[nt][r] *= alpha[r];
      bf16x16 bv = pack_a(*(const bf16x8*)&lVt[(nt * 16 + lcol) * 32 + kh * 8], *(const bf16x8*)&lVt[(nt * 16 + lcol) * 32 + kh * 8 + 16]);
      oacc[nt] = wmma_bf16(ap, bv, oacc[nt]);
    }
  }

  // write back token-major bf16 [4096, 1280] for the O-projection GEMM
#pragma unroll
  for (int r = 0; r < 8; ++r) {
    const float inv = 1.f / l_i[r];
    const size_t trow = (size_t)b * SEQ + q0 + wv * 16 + lrow + r;
#pragma unroll
    for (int nt = 0; nt < 4; ++nt)
      out[trow * DMODEL + h * HDIM + nt * 16 + lcol] = (__bf16)(oacc[nt][r] * inv);
  }
}

// ---------------- host-side launch ----------------
extern "C" void kernel_launch(void* const* d_in, const int* in_sizes, int n_in,
                              void* d_out, int out_size, void* d_ws, size_t ws_size,
                              hipStream_t stream) {
  const float* x     = (const float*)d_in[0];
  const unsigned char* amask = (const unsigned char*)d_in[1];
  const float* ln1_w = (const float*)d_in[2];
  const float* ln1_b = (const float*)d_in[3];
  const float* q_w   = (const float*)d_in[4];
  const float* q_b   = (const float*)d_in[5];
  const float* k_w   = (const float*)d_in[6];
  const float* k_b   = (const float*)d_in[7];
  const float* v_w   = (const float*)d_in[8];
  const float* v_b   = (const float*)d_in[9];
  const float* o_w   = (const float*)d_in[10];
  const float* o_b   = (const float*)d_in[11];
  const float* ln2_w = (const float*)d_in[12];
  const float* ln2_b = (const float*)d_in[13];
  const float* fc1_w = (const float*)d_in[14];
  const float* fc1_b = (const float*)d_in[15];
  const float* fc2_w = (const float*)d_in[16];
  const float* fc2_b = (const float*)d_in[17];

  // workspace bump allocator
  char* p = (char*)d_ws;
  auto alloc = [&](size_t bytes) -> char* {
    char* r = p;
    p += (bytes + 255) & ~(size_t)255;
    return r;
  };
  const size_t DD = (size_t)DMODEL * DMODEL;   // 1,638,400
  const size_t DF = (size_t)FFDIM * DMODEL;    // 6,553,600
  const size_t TD = (size_t)NTOK * DMODEL;     // 5,242,880
  const size_t TF = (size_t)NTOK * FFDIM;      // 20,971,520
  __bf16* Wq  = (__bf16*)alloc(DD * 2);
  __bf16* Wk  = (__bf16*)alloc(DD * 2);
  __bf16* Wv  = (__bf16*)alloc(DD * 2);
  __bf16* Wo  = (__bf16*)alloc(DD * 2);
  __bf16* W1  = (__bf16*)alloc(DF * 2);
  __bf16* W2  = (__bf16*)alloc(DF * 2);
  __bf16* hb  = (__bf16*)alloc(TD * 2);   // ln1 out, reused for ln2 out
  __bf16* qb  = (__bf16*)alloc(TD * 2);   // [b,H,n,64]
  __bf16* kb  = (__bf16*)alloc(TD * 2);
  __bf16* vb  = (__bf16*)alloc(TD * 2);
  __bf16* ab  = (__bf16*)alloc(TD * 2);   // attention out, token-major
  float*  x1  = (float*) alloc(TD * 4);   // x + attn projection (residual 1)
  __bf16* ff1 = (__bf16*)alloc(TF * 2);   // gelu(fc1) out

  const dim3 blk256(256);
  // 1) convert weights to bf16
  cvt_bf16_kernel<<<(int)(DD / 256), blk256, 0, stream>>>(q_w,  Wq, (int)DD);
  cvt_bf16_kernel<<<(int)(DD / 256), blk256, 0, stream>>>(k_w,  Wk, (int)DD);
  cvt_bf16_kernel<<<(int)(DD / 256), blk256, 0, stream>>>(v_w,  Wv, (int)DD);
  cvt_bf16_kernel<<<(int)(DD / 256), blk256, 0, stream>>>(o_w,  Wo, (int)DD);
  cvt_bf16_kernel<<<(int)(DF / 256), blk256, 0, stream>>>(fc1_w, W1, (int)DF);
  cvt_bf16_kernel<<<(int)(DF / 256), blk256, 0, stream>>>(fc2_w, W2, (int)DF);

  // 2) ln1
  ln_bf16_kernel<<<NTOK, blk256, 0, stream>>>(x, ln1_w, ln1_b, hb);

  // 3) QKV projections into head-major layout
  const dim3 gD(DMODEL / 128, NTOK / 128);
  gemm_kernel<1, 0, 1, 0><<<gD, blk256, 0, stream>>>(hb, Wq, q_b, nullptr, qb, DMODEL, DMODEL);
  gemm_kernel<1, 0, 1, 0><<<gD, blk256, 0, stream>>>(hb, Wk, k_b, nullptr, kb, DMODEL, DMODEL);
  gemm_kernel<1, 0, 1, 0><<<gD, blk256, 0, stream>>>(hb, Wv, v_b, nullptr, vb, DMODEL, DMODEL);

  // 4) attention
  attn_kernel<<<dim3(4 * NHEAD, SEQ / 64), dim3(128), 0, stream>>>(qb, kb, vb, amask, ab);

  // 5) O projection + residual -> x1 (f32)
  gemm_kernel<0, 0, 0, 1><<<gD, blk256, 0, stream>>>(ab, Wo, o_b, x, x1, DMODEL, DMODEL);

  // 6) ln2
  ln_bf16_kernel<<<NTOK, blk256, 0, stream>>>(x1, ln2_w, ln2_b, hb);

  // 7) fc1 + gelu
  const dim3 gF(FFDIM / 128, NTOK / 128);
  gemm_kernel<1, 1, 0, 0><<<gF, blk256, 0, stream>>>(hb, W1, fc1_b, nullptr, ff1, FFDIM, DMODEL);

  // 8) fc2 + residual -> out (f32)
  gemm_kernel<0, 0, 0, 1><<<gD, blk256, 0, stream>>>(ff1, W2, fc2_b, x1, (float*)d_out,
                                                     DMODEL, FFDIM);
}
